// QuantizedLinear_65678639891121
// MI455X (gfx1250) — 8-bit matrix operand order repaired and checked by reading, compiled, not run on hardware
//
#include <hip/hip_runtime.h>

typedef int v8i __attribute__((ext_vector_type(8)));

#define M_DIM 16384
#define N_DIM 4096
#define K_DIM 4096

__device__ __forceinline__ int quant1(float v, float s, int zp) {
    int q = (int)rintf(v / s) + zp;
    return q < -128 ? -128 : (q > 127 ? 127 : q);
}

__global__ __launch_bounds__(256) void quantize_x_kernel(
    const float* __restrict__ x,
    const float* __restrict__ act_scale_p,
    const int*   __restrict__ zp_p,
    signed char* __restrict__ qx) {
    const float s  = act_scale_p[0];
    const int   zp = zp_p[0];
    const size_t tid = (size_t)blockIdx.x * blockDim.x + threadIdx.x;
    const float4* xin = (const float4*)x;
    int packed[4];
#pragma unroll
    for (int j = 0; j < 4; ++j) {
        float4 v = xin[tid * 4 + j];
        int q0 = quant1(v.x, s, zp);
        int q1 = quant1(v.y, s, zp);
        int q2 = quant1(v.z, s, zp);
        int q3 = quant1(v.w, s, zp);
        packed[j] = (q0 & 0xFF) | ((q1 & 0xFF) << 8) |
                    ((q2 & 0xFF) << 16) | ((q3 & 0xFF) << 24);
    }
    int4 o; o.x = packed[0]; o.y = packed[1]; o.z = packed[2]; o.w = packed[3];
    ((int4*)qx)[tid] = o;
}

__global__ __launch_bounds__(256) void quantize_w_kernel(
    const float* __restrict__ w,
    const float* __restrict__ wscale,
    signed char* __restrict__ qw,
    int*         __restrict__ wsum) {
    __shared__ int red[256];
    const int n = blockIdx.x;
    const float s = wscale[n];
    const float4* wrow = (const float4*)(w + (size_t)n * K_DIM);
    int* qrow = (int*)(qw + (size_t)n * K_DIM);
    int sum = 0;
#pragma unroll
    for (int it = 0; it < K_DIM / 4 / 256; ++it) {
        int idx = it * 256 + threadIdx.x;
        float4 v = wrow[idx];
        int q0 = quant1(v.x, s, 0);
        int q1 = quant1(v.y, s, 0);
        int q2 = quant1(v.z, s, 0);
        int q3 = quant1(v.w, s, 0);
        sum += q0 + q1 + q2 + q3;
        qrow[idx] = (q0 & 0xFF) | ((q1 & 0xFF) << 8) |
                    ((q2 & 0xFF) << 16) | ((q3 & 0xFF) << 24);
    }
    red[threadIdx.x] = sum;
    __syncthreads();
#pragma unroll
    for (int off = 128; off > 0; off >>= 1) {
        if ((int)threadIdx.x < off) red[threadIdx.x] += red[threadIdx.x + off];
        __syncthreads();
    }
    if (threadIdx.x == 0) wsum[n] = red[0];
}

__global__ __launch_bounds__(256) void gemm_iu8_kernel(
    const signed char* __restrict__ qx,
    const signed char* __restrict__ qw,
    const int*   __restrict__ wsum,
    const float* __restrict__ wscale,
    const float* __restrict__ bias,
    const float* __restrict__ act_scale_p,
    const int*   __restrict__ zp_p,
    float*       __restrict__ out) {
    const int lane = threadIdx.x & 31;
    const int wave = threadIdx.x >> 5;
    const int wm = wave >> 2;
    const int wn = wave & 3;
    const int baseM = blockIdx.y * 128 + wm * 64;
    const int baseN = blockIdx.x * 256 + wn * 64;
    const int half = lane >> 4;
    const int l16  = lane & 15;

    v8i acc[4][4];
    {
        v8i zero = {};
#pragma unroll
        for (int mt = 0; mt < 4; ++mt)
#pragma unroll
            for (int nt = 0; nt < 4; ++nt) acc[mt][nt] = zero;
    }

    const unsigned long long* arow[4];
#pragma unroll
    for (int mt = 0; mt < 4; ++mt)
        arow[mt] = (const unsigned long long*)
            (qx + (size_t)(baseM + mt * 16 + l16) * K_DIM + half * 16);
    const int4* brow[4];
#pragma unroll
    for (int nt = 0; nt < 4; ++nt)
        brow[nt] = (const int4*)
            (qw + (size_t)(baseN + nt * 16 + l16) * K_DIM + half * 16);

    for (int k0 = 0; k0 < K_DIM; k0 += 64) {
        if (k0 + 64 < K_DIM) {
            __builtin_prefetch((const char*)arow[0] + k0 + 64, 0, 3);
            __builtin_prefetch((const char*)brow[0] + k0 + 64, 0, 3);
        }
        v8i a[4];
#pragma unroll
        for (int mt = 0; mt < 4; ++mt) {
            const unsigned long long* p = arow[mt] + (k0 >> 3);
#pragma unroll
            for (int g = 0; g < 4; ++g) {
                unsigned long long t = p[g + (g & 2)];
                a[mt][2 * g]     = (int)(unsigned)(t & 0xffffffffull);
                a[mt][2 * g + 1] = (int)(unsigned)(t >> 32);
            }
        }
        v8i b[4];
#pragma unroll
        for (int nt = 0; nt < 4; ++nt) {
            const int4* p = brow[nt] + (k0 >> 4);
            int4 t0 = p[0];
            int4 t1 = p[2];
            b[nt][0] = t0.x; b[nt][1] = t0.y; b[nt][2] = t0.z; b[nt][3] = t0.w;
            b[nt][4] = t1.x; b[nt][5] = t1.y; b[nt][6] = t1.z; b[nt][7] = t1.w;
        }
#pragma unroll
        for (int mt = 0; mt < 4; ++mt)
#pragma unroll
            for (int nt = 0; nt < 4; ++nt)
                acc[mt][nt] = __builtin_amdgcn_wmma_i32_16x16x64_iu8(
                    true, a[mt], true, b[nt], acc[mt][nt], false, false);
    }

    const float s_act = act_scale_p[0];
    const int   zp    = zp_p[0];
#pragma unroll
    for (int nt = 0; nt < 4; ++nt) {
        const int col   = baseN + nt * 16 + l16;
        const float scl = s_act * wscale[col];
        const float bi  = bias[col];
        const int corr  = zp * wsum[col];
#pragma unroll
        for (int mt = 0; mt < 4; ++mt) {
            const int rowB = baseM + mt * 16 + half * 8;
#pragma unroll
            for (int r = 0; r < 8; ++r)
                out[(size_t)(rowB + r) * N_DIM + col] =
                    (float)(acc[mt][nt][r] - corr) * scl + bi;
        }
    }
}

extern "C" void kernel_launch(void* const* d_in, const int* in_sizes, int n_in,
                              void* d_out, int out_size, void* d_ws, size_t ws_size,
                              hipStream_t stream) {
    const float* x         = (const float*)d_in[0];
    const float* weight    = (const float*)d_in[1];
    const float* bias      = (const float*)d_in[2];
    const float* act_scale = (const float*)d_in[3];
    const int*   act_zp    = (const int*)d_in[4];
    const float* wscale    = (const float*)d_in[5];
    float* out = (float*)d_out;

    signed char* qx = (signed char*)d_ws;
    signed char* qw = qx + (size_t)M_DIM * K_DIM;
    int* wsum = (int*)(qw + (size_t)N_DIM * K_DIM);

    quantize_x_kernel<<<((size_t)M_DIM * K_DIM) / 16 / 256, 256, 0, stream>>>(
        x, act_scale, act_zp, qx);

    quantize_w_kernel<<<N_DIM, 256, 0, stream>>>(weight, wscale, qw, wsum);

    dim3 grid(N_DIM / 256, M_DIM / 128);
    gemm_iu8_kernel<<<grid, 256, 0, stream>>>(
        qx, qw, wsum, wscale, bias, act_scale, act_zp, out);
}
